// RotaryPositionMultiHeadAttention_21174188769412
// MI455X (gfx1250) — matrix operand order repaired and checked by reading, not compiled, not run on hardware
//
#include <hip/hip_runtime.h>
#include <stdint.h>

// ---------------------------------------------------------------------------
// RotaryPositionMultiHeadAttention for MI455X (gfx1250, wave32, WMMA)
//   B=2, T=2048, H=16, DK=64, F=1024
// Pipeline:
//   1) cvt weights f32->bf16 (wq,bq pre-scaled by 1/sqrt(DK))
//   2) RoPE(q,k) + cvt(q,k,v) -> bf16
//   3) Q = qr@wq^T+bq -> [b,h,t,d] bf16 ; K likewise ; V = v@wv^T+bv -> [b,h,d,t]
//   4) flash attention per 16-row q tile (online softmax), O -> [bt,f] bf16
//   5) out = O@wo^T+bo -> f32 (d_out)
// All GEMMs use v_wmma_f32_16x16x32_bf16 (f32 accumulate).
// ---------------------------------------------------------------------------

#define kH  16
#define kDK 64
#define kF  1024
#define kB  2
#define kT  2048
#define kM  (kB * kT) // 4096

typedef __attribute__((ext_vector_type(16))) __bf16 v16bf;
typedef __attribute__((ext_vector_type(8)))  __bf16 v8bf;
typedef __attribute__((ext_vector_type(8)))  float  v8f;

static __device__ __forceinline__ v8f wmma_bf16(v16bf a, v16bf b, v8f c) {
  // D = A(16x32) * B(32x16) + C(16x16), f32 accumulate
  return __builtin_amdgcn_wmma_f32_16x16x32_bf16(false, a, false, b, (short)0, c,
                                                 false, false);
}

static __device__ __forceinline__ v16bf join16(v8bf lo, v8bf hi) {
  union { v16bf v; v8bf h[2]; } u;
  u.h[0] = lo;
  u.h[1] = hi;
  return u.v;
}

// ---------------------------------------------------------------------------
// f32 -> bf16 convert with optional scale (used for the 4 weight matrices)
// ---------------------------------------------------------------------------
__global__ void cvt_bf16_kernel(const float* __restrict__ src,
                                unsigned short* __restrict__ dst_raw,
                                int n, float scale) {
  __bf16* dst = (__bf16*)dst_raw;
  int i = blockIdx.x * blockDim.x + threadIdx.x;
  const int stride = gridDim.x * blockDim.x;
  for (; i < n; i += stride) dst[i] = (__bf16)(src[i] * scale);
}

// ---------------------------------------------------------------------------
// RoPE on q,k (per head of 64, rotate-half) + cvt q,k,v to bf16.
// grid = B*T blocks, 256 threads, 4 elements/thread.
// ---------------------------------------------------------------------------
__global__ void rope_cvt_kernel(const float* __restrict__ q,
                                const float* __restrict__ k,
                                const float* __restrict__ v,
                                const float* __restrict__ cosb,
                                const float* __restrict__ sinb,
                                unsigned short* __restrict__ qr_raw,
                                unsigned short* __restrict__ kr_raw,
                                unsigned short* __restrict__ vb_raw) {
  __bf16* qr = (__bf16*)qr_raw;
  __bf16* kr = (__bf16*)kr_raw;
  __bf16* vb = (__bf16*)vb_raw;
  const int row = blockIdx.x;            // b*T + t
  const int t   = row & (kT - 1);
  const int f0  = threadIdx.x * 4;
  const size_t base = (size_t)row * kF;
#pragma unroll
  for (int j = 0; j < 4; ++j) {
    const int f = f0 + j;
    const int d = f & (kDK - 1);
    const float c = cosb[t * kDK + d];
    const float s = sinb[t * kDK + d];
    const int   ro = (d < kDK / 2) ? (kDK / 2) : -(kDK / 2);
    const float sg = (d < kDK / 2) ? -1.0f : 1.0f;
    qr[base + f] = (__bf16)(q[base + f] * c + sg * q[base + f + ro] * s);
    kr[base + f] = (__bf16)(k[base + f] * c + sg * k[base + f + ro] * s);
    vb[base + f] = (__bf16)(v[base + f]);
  }
}

// ---------------------------------------------------------------------------
// Y = X(M=4096 x K=1024) @ W(N=1024 x K=1024)^T + bias
// 4 waves/block; each wave owns one 16x16 output tile, K-loop step 32.
// mode 0: store bf16 -> [b,h,t,d]    (Q, K)
// mode 1: store bf16 -> [b,h,d,t]    (V, transposed for PV B-fragments)
// mode 2: store f32  -> row-major    (final output)
// ---------------------------------------------------------------------------
__global__ __launch_bounds__(128)
void gemm_bf16_kernel(const unsigned short* __restrict__ X_raw,
                      const unsigned short* __restrict__ W_raw,
                      const float* __restrict__ bias, float bias_scale,
                      unsigned short* __restrict__ outb_raw,
                      float* __restrict__ outf, int mode) {
  const __bf16* X = (const __bf16*)X_raw;
  const __bf16* W = (const __bf16*)W_raw;
  __bf16* outb = (__bf16*)outb_raw;

  const int lane = threadIdx.x & 31;
  const int wave = threadIdx.x >> 5;
  const int n16  = lane & 15;   // A: row m / B: col n / C: col n
  const int hb   = lane >> 4;   // half-wave selector (K / M-half per ISA layout)

  const int m_tile = blockIdx.x & (kM / 16 - 1); // 256 m-tiles
  const int n_blk  = blockIdx.x / (kM / 16);     // 16 n-blocks of 64 cols
  const int n0     = (n_blk * 4 + wave) * 16;
  const int row0   = m_tile * 16;

  const __bf16* xrow = X + (size_t)(row0 + n16) * kF;
  const __bf16* wrow = W + (size_t)(n0 + n16) * kF;

  v8f acc = {0.f, 0.f, 0.f, 0.f, 0.f, 0.f, 0.f, 0.f};
#pragma unroll 4
  for (int k0 = 0; k0 < kF; k0 += 32) {
    v16bf a = join16(*(const v8bf*)(xrow + k0 + 8 * hb),
                     *(const v8bf*)(xrow + k0 + 16 + 8 * hb));
    v16bf b = join16(*(const v8bf*)(wrow + k0 + 8 * hb),
                     *(const v8bf*)(wrow + k0 + 16 + 8 * hb));
    acc = wmma_bf16(a, b, acc);
  }

  const int   nglob = n0 + n16;
  const float bv    = bias[nglob] * bias_scale;

  if (mode == 2) {
#pragma unroll
    for (int r = 0; r < 8; ++r) {
      const int m = r + 8 * hb;
      outf[(size_t)(row0 + m) * kF + nglob] = acc[r] + bv;
    }
  } else {
    const int hh = nglob >> 6;
    const int d  = nglob & (kDK - 1);
    const int bb = row0 >> 11;       // row0 / kT  (16-row tile never crosses b)
    const int tb = row0 & (kT - 1);
    if (mode == 0) {
#pragma unroll
      for (int r = 0; r < 8; ++r) {
        const int m = r + 8 * hb;
        outb[((size_t)(bb * kH + hh) * kT + tb + m) * kDK + d] =
            (__bf16)(acc[r] + bv);
      }
    } else {
      v8bf pk;
#pragma unroll
      for (int r = 0; r < 8; ++r) pk[r] = (__bf16)(acc[r] + bv);
      __bf16* dst = outb + ((size_t)(bb * kH + hh) * kDK + d) * kT + tb + 8 * hb;
      *(v8bf*)dst = pk;  // 8 consecutive t -> one 16B store
    }
  }
}

// ---------------------------------------------------------------------------
// Flash attention. Q,K in [b,h,t,d] bf16 (1/8 score scale folded into Q),
// V in [b,h,d,t] bf16. One wave per 16-row q tile; 4 waves/block share (b,h).
// Per 32-key step: 4 score WMMAs, online softmax (shfl_xor reductions over the
// 16-lane half-groups), P transposed C-layout -> A-frag via LDS, 4 PV WMMAs.
// Output O -> [bt, f] bf16.
// ---------------------------------------------------------------------------
__global__ __launch_bounds__(128)
void attn_kernel(const unsigned short* __restrict__ Qh_raw,
                 const unsigned short* __restrict__ Kh_raw,
                 const unsigned short* __restrict__ Vt_raw,
                 unsigned short* __restrict__ O_raw) {
  const __bf16* Qh = (const __bf16*)Qh_raw;
  const __bf16* Kh = (const __bf16*)Kh_raw;
  const __bf16* Vt = (const __bf16*)Vt_raw;
  __bf16* O = (__bf16*)O_raw;

  __shared__ __align__(64) __bf16 pbuf[4][16][32];

  const int lane = threadIdx.x & 31;
  const int wave = threadIdx.x >> 5;
  const int n16  = lane & 15;
  const int hb   = lane >> 4;

  const int nchunk = kT / 64;                 // 32 q-chunks of 64 rows
  const int bh     = blockIdx.x / nchunk;     // b*H + h
  const int chunk  = blockIdx.x % nchunk;
  const int b      = bh / kH;
  const int h      = bh % kH;
  const int qt     = chunk * 64 + wave * 16;  // q tile start row

  const size_t bhT = (size_t)bh * kT;

  // Q A-fragments for d0 = 0 and d0 = 32 (held for the whole key loop)
  const __bf16* qrow = Qh + (bhT + qt + n16) * kDK;
  const v16bf aq0 = *(const v16bf*)(qrow + 16 * hb);
  const v16bf aq1 = *(const v16bf*)(qrow + 32 + 16 * hb);

  float mrun[8], lrun[8];
  v8f oacc[4];
#pragma unroll
  for (int r = 0; r < 8; ++r) { mrun[r] = -3.0e38f; lrun[r] = 0.0f; }
#pragma unroll
  for (int tt = 0; tt < 4; ++tt)
    oacc[tt] = (v8f){0.f, 0.f, 0.f, 0.f, 0.f, 0.f, 0.f, 0.f};

  const v8f vzero = {0.f, 0.f, 0.f, 0.f, 0.f, 0.f, 0.f, 0.f};

  for (int j0 = 0; j0 < kT; j0 += 32) {
    // ---- scores: S(16x32) = Qscaled(16x64) @ K^T, two 16-key tiles ----
    const __bf16* k0row = Kh + (bhT + j0 + n16) * kDK;
    const __bf16* k1row = k0row + 16 * kDK;
    const v16bf kb0l = *(const v16bf*)(k0row + 16 * hb);
    const v16bf kb0h = *(const v16bf*)(k0row + 32 + 16 * hb);
    const v16bf kb1l = *(const v16bf*)(k1row + 16 * hb);
    const v16bf kb1h = *(const v16bf*)(k1row + 32 + 16 * hb);

    v8f s0 = wmma_bf16(aq1, kb0h, wmma_bf16(aq0, kb0l, vzero));
    v8f s1 = wmma_bf16(aq1, kb1h, wmma_bf16(aq0, kb1l, vzero));

    // ---- online softmax over the 32 new columns ----
    float rmax[8];
#pragma unroll
    for (int r = 0; r < 8; ++r) rmax[r] = fmaxf(s0[r], s1[r]);
#pragma unroll
    for (int mk = 1; mk < 16; mk <<= 1) {
#pragma unroll
      for (int r = 0; r < 8; ++r)
        rmax[r] = fmaxf(rmax[r], __shfl_xor(rmax[r], mk, 32));
    }

    float p0[8], p1[8], corr[8], rsum[8];
#pragma unroll
    for (int r = 0; r < 8; ++r) {
      const float mnew = fmaxf(mrun[r], rmax[r]);
      corr[r] = __expf(mrun[r] - mnew);
      mrun[r] = mnew;
      p0[r] = __expf(s0[r] - mnew);
      p1[r] = __expf(s1[r] - mnew);
      rsum[r] = p0[r] + p1[r];
    }
#pragma unroll
    for (int mk = 1; mk < 16; mk <<= 1) {
#pragma unroll
      for (int r = 0; r < 8; ++r) rsum[r] += __shfl_xor(rsum[r], mk, 32);
    }
#pragma unroll
    for (int r = 0; r < 8; ++r) lrun[r] = lrun[r] * corr[r] + rsum[r];
#pragma unroll
    for (int tt = 0; tt < 4; ++tt)
#pragma unroll
      for (int r = 0; r < 8; ++r) oacc[tt][r] *= corr[r];

    // ---- transpose P (C layout) -> A fragment via per-wave LDS tile ----
#pragma unroll
    for (int r = 0; r < 8; ++r) {
      const int m = r + 8 * hb;
      pbuf[wave][m][n16]      = (__bf16)p0[r];
      pbuf[wave][m][16 + n16] = (__bf16)p1[r];
    }
    // DS ops from one wave are in-order; pin compiler ordering + wait.
    asm volatile("s_wait_dscnt 0" ::: "memory");

    const __bf16* prow = &pbuf[wave][n16][0];
    const v16bf pa = *(const v16bf*)(prow + 16 * hb);

    // ---- O(16x64) += P(16x32) @ V(32x64), V read contiguous from [b,h,d,t]
#pragma unroll
    for (int tt = 0; tt < 4; ++tt) {
      const __bf16* vrow =
          Vt + ((size_t)bh * kDK + tt * 16 + n16) * kT + j0 + 16 * hb;
      const v16bf vb = *(const v16bf*)vrow;
      oacc[tt] = wmma_bf16(pa, vb, oacc[tt]);
    }
  }

  // ---- epilogue: O /= l, store bf16 to [bt, f] ----
#pragma unroll
  for (int tt = 0; tt < 4; ++tt) {
#pragma unroll
    for (int r = 0; r < 8; ++r) {
      const int m = r + 8 * hb;
      const float val = oacc[tt][r] / lrun[r];
      O[(size_t)(b * kT + qt + m) * kF + h * kDK + tt * 16 + n16] = (__bf16)val;
    }
  }
}

// ---------------------------------------------------------------------------
// Host launcher
// ---------------------------------------------------------------------------
extern "C" void kernel_launch(void* const* d_in, const int* in_sizes, int n_in,
                              void* d_out, int out_size, void* d_ws,
                              size_t ws_size, hipStream_t stream) {
  (void)in_sizes; (void)n_in; (void)out_size; (void)ws_size;

  const float* q    = (const float*)d_in[0];
  const float* k    = (const float*)d_in[1];
  const float* v    = (const float*)d_in[2];
  const float* cosb = (const float*)d_in[3];
  const float* sinb = (const float*)d_in[4];
  const float* wq   = (const float*)d_in[5];
  const float* wk   = (const float*)d_in[6];
  const float* wv   = (const float*)d_in[7];
  const float* wo   = (const float*)d_in[8];
  const float* bq   = (const float*)d_in[9];
  const float* bk   = (const float*)d_in[10];
  const float* bv   = (const float*)d_in[11];
  const float* bo   = (const float*)d_in[12];
  float* out        = (float*)d_out;

  char* ws = (char*)d_ws;
  size_t off = 0;
  auto alloc = [&](size_t bytes) -> unsigned short* {
    unsigned short* p = (unsigned short*)(ws + off);
    off += (bytes + 255) & ~(size_t)255;
    return p;
  };

  const size_t ROWB = (size_t)kM * kF * 2;  // 8 MB per (B,T,F) bf16 tensor
  const size_t WB   = (size_t)kF * kF * 2;  // 2 MB per weight bf16
  unsigned short* q_r  = alloc(ROWB);
  unsigned short* k_r  = alloc(ROWB);
  unsigned short* v_b  = alloc(ROWB);
  unsigned short* wq_b = alloc(WB);
  unsigned short* wk_b = alloc(WB);
  unsigned short* wv_b = alloc(WB);
  unsigned short* wo_b = alloc(WB);
  unsigned short* Qh   = alloc(ROWB);
  unsigned short* Kh   = alloc(ROWB);
  unsigned short* Vt   = alloc(ROWB);
  unsigned short* Obf  = alloc(ROWB);

  const float inv_sqrt_d = 0.125f;  // 1/sqrt(DK); folded into wq & bq

  cvt_bf16_kernel<<<512, 256, 0, stream>>>(wq, wq_b, kF * kF, inv_sqrt_d);
  cvt_bf16_kernel<<<512, 256, 0, stream>>>(wk, wk_b, kF * kF, 1.0f);
  cvt_bf16_kernel<<<512, 256, 0, stream>>>(wv, wv_b, kF * kF, 1.0f);
  cvt_bf16_kernel<<<512, 256, 0, stream>>>(wo, wo_b, kF * kF, 1.0f);

  rope_cvt_kernel<<<kM, 256, 0, stream>>>(q, k, v, cosb, sinb, q_r, k_r, v_b);

  const int gemm_grid = (kM / 16) * (kF / 64);  // 4096 blocks, 4 waves each
  gemm_bf16_kernel<<<gemm_grid, 128, 0, stream>>>(q_r, wq_b, bq, inv_sqrt_d,
                                                  Qh, nullptr, 0);
  gemm_bf16_kernel<<<gemm_grid, 128, 0, stream>>>(k_r, wk_b, bk, 1.0f,
                                                  Kh, nullptr, 0);
  gemm_bf16_kernel<<<gemm_grid, 128, 0, stream>>>(v_b, wv_b, bv, 1.0f,
                                                  Vt, nullptr, 1);

  attn_kernel<<<kB * kH * (kT / 64), 128, 0, stream>>>(Qh, Kh, Vt, Obf);

  gemm_bf16_kernel<<<gemm_grid, 128, 0, stream>>>(Obf, wo_b, bo, 1.0f,
                                                  nullptr, out, 2);
}
